// BiLSTM_CRF_41205916238362
// MI455X (gfx1250) — matrix operand order repaired and checked by reading, not compiled, not run on hardware
//
#include <hip/hip_runtime.h>
#include <stdint.h>

#define B_ 128
#define S_ 256
#define E_ 256
#define H_ 256
#define T_ 22   // TAG + START + STOP
// START = 20, STOP = 21

typedef __attribute__((ext_vector_type(16))) __bf16 v16bf;
typedef __attribute__((ext_vector_type(8)))  float  v8f;

union FragAB { v16bf v; uint4 q[2]; };
union FragC  { v8f  v; float f[8]; };

__device__ __forceinline__ uint16_t f2bf(float f) {
    uint32_t x = __float_as_uint(f);
    uint32_t r = x + 0x7FFFu + ((x >> 16) & 1u);   // round-to-nearest-even
    return (uint16_t)(r >> 16);
}
__device__ __forceinline__ float bf2f(uint16_t h) {
    return __uint_as_float(((uint32_t)h) << 16);
}
__device__ __forceinline__ float sigmoidf_(float x) {
    return 1.0f / (1.0f + __expf(-x));
}

// ---------------------------------------------------------------- prep kernels
__global__ void k_f32_to_bf16(const float* __restrict__ src,
                              uint16_t* __restrict__ dst, int n) {
    int i = blockIdx.x * blockDim.x + threadIdx.x;
    if (i < n) dst[i] = f2bf(src[i]);
}

__global__ void k_bias_sum(const float* __restrict__ a, const float* __restrict__ b,
                           float* __restrict__ o, int n) {
    int i = blockIdx.x * blockDim.x + threadIdx.x;
    if (i < n) o[i] = a[i] + b[i];
}

// embedding gather -> bf16 activations (B,S,E)
__global__ void k_embed(const float* __restrict__ emb, const int* __restrict__ words,
                        uint16_t* __restrict__ x, int n) {
    int i = blockIdx.x * blockDim.x + threadIdx.x;
    if (i >= n) return;
    int e  = i & (E_ - 1);
    int bs = i >> 8;                       // E_ == 256
    int w  = words[bs];
    x[i] = f2bf(emb[(size_t)w * E_ + e]);
}

// ---------------------------------------------------------------- fused LSTM scan
// grid = 16 blocks: blockIdx = (batch_tile<<1)|dir; block = 256 threads (8 waves)
// Per step: gates(16x1024) = [x_t ; h] x [Wih ; Whh]^T + bias, all via
// v_wmma_f32_16x16x32_bf16, then elementwise cell update. h lives in LDS (bf16).
__global__ void __launch_bounds__(256)
k_lstm_scan(const uint16_t* __restrict__ xin, int inDim,
            const uint16_t* __restrict__ wih,   // (2, 1024, inDim) bf16 row-major
            const uint16_t* __restrict__ whh,   // (2, 1024, 256)   bf16 row-major
            const float*    __restrict__ bsum,  // (2, 1024) = b_ih + b_hh
            uint16_t*       __restrict__ out,   // (B, S, 512) bf16; dir writes half
            float*          __restrict__ gates_ws) // 16 * 16*1024 f32
{
    const int dir  = blockIdx.x & 1;
    const int bt   = blockIdx.x >> 1;
    const int b0   = bt * 16;
    const int tid  = threadIdx.x;
    const int lane = tid & 31;
    const int wave = tid >> 5;
    const int l15  = lane & 15;
    const int hi   = (lane >> 4) & 1;

    float* __restrict__ g = gates_ws + (size_t)blockIdx.x * (16 * 1024);
    const uint16_t* __restrict__ wihD = wih + (size_t)dir * 1024 * inDim;
    const uint16_t* __restrict__ whhD = whh + (size_t)dir * 1024 * H_;
    const float*    __restrict__ bsD  = bsum + dir * 1024;

    __shared__ __align__(16) uint16_t hsh[16 * H_];   // 8 KB: h for this batch tile

    for (int i = tid; i < 16 * H_; i += 256) hsh[i] = 0;
    float creg[16];
#pragma unroll
    for (int k = 0; k < 16; ++k) creg[k] = 0.f;
    __syncthreads();

    for (int t = 0; t < S_; ++t) {
        const int ts = dir ? (S_ - 1 - t) : t;

        // A-fragment row bases (16-bit A layout: lo lanes K 0-7/16-23, hi lanes K 8-15/24-31)
        const uint16_t* __restrict__ arow =
            xin + ((size_t)(b0 + l15) * S_ + ts) * inDim + hi * 8;
        const uint16_t* __restrict__ hrow = hsh + l15 * H_ + hi * 8;

        for (int tile = wave; tile < 64; tile += 8) {
            const int n0 = tile * 16;
            FragC c;
            const float bias = bsD[n0 + l15];
#pragma unroll
            for (int r = 0; r < 8; ++r) c.f[r] = bias;

            // input contribution: x_t @ Wih^T
            const uint16_t* __restrict__ bi = wihD + (size_t)(n0 + l15) * inDim + hi * 8;
            for (int k0 = 0; k0 < inDim; k0 += 32) {
                FragAB a, b;
                a.q[0] = *reinterpret_cast<const uint4*>(arow + k0);
                a.q[1] = *reinterpret_cast<const uint4*>(arow + k0 + 16);
                b.q[0] = *reinterpret_cast<const uint4*>(bi + k0);
                b.q[1] = *reinterpret_cast<const uint4*>(bi + k0 + 16);
                c.v = __builtin_amdgcn_wmma_f32_16x16x32_bf16(
                        false, a.v, false, b.v, (short)0, c.v, false, false);
            }
            // recurrent contribution: h @ Whh^T (A from LDS)
            const uint16_t* __restrict__ bh = whhD + (size_t)(n0 + l15) * H_ + hi * 8;
            for (int k0 = 0; k0 < H_; k0 += 32) {
                FragAB a, b;
                a.q[0] = *reinterpret_cast<const uint4*>(hrow + k0);
                a.q[1] = *reinterpret_cast<const uint4*>(hrow + k0 + 16);
                b.q[0] = *reinterpret_cast<const uint4*>(bh + k0);
                b.q[1] = *reinterpret_cast<const uint4*>(bh + k0 + 16);
                c.v = __builtin_amdgcn_wmma_f32_16x16x32_bf16(
                        false, a.v, false, b.v, (short)0, c.v, false, false);
            }
            // C/D layout: VGPR r -> (m = r + 8*hi, n = n0 + l15)
#pragma unroll
            for (int r = 0; r < 8; ++r)
                g[(size_t)(r + hi * 8) * 1024 + n0 + l15] = c.f[r];
        }
        __threadfence();
        __syncthreads();

        // elementwise LSTM cell update: 16 (b,j) pairs per thread
#pragma unroll
        for (int k = 0; k < 16; ++k) {
            const int idx = k * 256 + tid;
            const int bl  = idx >> 8;     // local batch row 0..15
            const int j   = idx & 255;    // hidden index 0..255
            const float gi = g[bl * 1024 + j];
            const float gf = g[bl * 1024 + 256 + j];
            const float gg = g[bl * 1024 + 512 + j];
            const float go = g[bl * 1024 + 768 + j];
            const float cc = sigmoidf_(gf) * creg[k] + sigmoidf_(gi) * tanhf(gg);
            creg[k] = cc;
            const float hh = sigmoidf_(go) * tanhf(cc);
            const uint16_t hb = f2bf(hh);
            hsh[bl * H_ + j] = hb;
            out[((size_t)(b0 + bl) * S_ + ts) * (2 * H_) + dir * H_ + j] = hb;
        }
        __syncthreads();
    }
}

// ---------------------------------------------------------------- LN + projection
__global__ void __launch_bounds__(256)
k_ln_proj(const uint16_t* __restrict__ xin,   // (B*S, 512) bf16
          const float* __restrict__ ln_g, const float* __restrict__ ln_b,
          const float* __restrict__ w_out, const float* __restrict__ b_out,
          float* __restrict__ feats)          // (B*S, 22)
{
    __shared__ float xs[512];
    __shared__ float r1[256];
    __shared__ float r2[256];
    const int tid = threadIdx.x;
    const size_t bs = blockIdx.x;
    const uint16_t* row = xin + bs * 512;
    const float v0 = bf2f(row[2 * tid]);
    const float v1 = bf2f(row[2 * tid + 1]);
    r1[tid] = v0 + v1;
    r2[tid] = v0 * v0 + v1 * v1;
    __syncthreads();
    for (int off = 128; off > 0; off >>= 1) {
        if (tid < off) { r1[tid] += r1[tid + off]; r2[tid] += r2[tid + off]; }
        __syncthreads();
    }
    const float mu  = r1[0] * (1.f / 512.f);
    const float var = r2[0] * (1.f / 512.f) - mu * mu;
    const float rs  = rsqrtf(var + 1e-5f);
    xs[2 * tid]     = (v0 - mu) * rs * ln_g[2 * tid]     + ln_b[2 * tid];
    xs[2 * tid + 1] = (v1 - mu) * rs * ln_g[2 * tid + 1] + ln_b[2 * tid + 1];
    __syncthreads();
    const int lane = tid & 31, wave = tid >> 5;
    for (int tt = wave; tt < T_; tt += 8) {
        const float* wr = w_out + tt * 512;
        float p = 0.f;
        for (int e = lane; e < 512; e += 32) p += xs[e] * wr[e];
        for (int off = 16; off > 0; off >>= 1) p += __shfl_xor(p, off);
        if (lane == 0) feats[bs * T_ + tt] = p + b_out[tt];
    }
}

// ---------------------------------------------------------------- CRF forward NLL
// one wave per sequence; lane j holds alpha[j]
__global__ void __launch_bounds__(32)
k_crf(const float* __restrict__ feats, const float* __restrict__ trans,
      const int* __restrict__ mask, const int* __restrict__ tags,
      float* __restrict__ nll)
{
    __shared__ float tr[T_ * T_];
    const int tid = threadIdx.x;
    for (int i = tid; i < T_ * T_; i += 32) tr[i] = trans[i];
    __syncthreads();

    const int b = blockIdx.x;
    const float* fb = feats + (size_t)b * S_ * T_;
    const int*   mb = mask  + b * S_;
    const int*   tg = tags  + b * S_;
    const int j = tid;
    const bool act = j < T_;

    float alpha = act ? (tr[20 * T_ + j] + fb[j]) : -1e30f;
    for (int s = 1; s < S_; ++s) {
        float m = -1e30f, acc = 0.f;
        for (int i = 0; i < T_; ++i) {
            const float ai = __shfl(alpha, i);
            const float v  = ai + (act ? tr[i * T_ + j] : 0.f);
            if (v > m) { acc = acc * __expf(m - v) + 1.f; m = v; }
            else       { acc += __expf(v - m); }
        }
        const float nw = m + __logf(acc) + fb[s * T_ + (act ? j : 0)];
        if (act && mb[s] > 0) alpha = nw;
    }
    // logZ = logsumexp_j(alpha_j + trans[j, STOP])
    float term = act ? (alpha + tr[j * T_ + 21]) : -1e30f;
    float m2 = term;
    for (int off = 16; off > 0; off >>= 1) m2 = fmaxf(m2, __shfl_xor(m2, off));
    float e2 = __expf(term - m2);
    for (int off = 16; off > 0; off >>= 1) e2 += __shfl_xor(e2, off);
    const float logZ = m2 + __logf(e2);

    // gold-path score
    float emit = 0.f, trs = 0.f; int cnt = 0;
    for (int s = tid; s < S_; s += 32) {
        const int mt = mb[s];
        cnt += mt;
        if (mt > 0) {
            emit += fb[s * T_ + tg[s]];
            if (s >= 1) trs += tr[tg[s - 1] * T_ + tg[s]];
        }
    }
    for (int off = 16; off > 0; off >>= 1) {
        emit += __shfl_xor(emit, off);
        trs  += __shfl_xor(trs,  off);
        cnt  += __shfl_xor(cnt,  off);
    }
    if (tid == 0) {
        trs += tr[20 * T_ + tg[0]];            // START -> tags[0]
        trs += tr[tg[cnt - 1] * T_ + 21];      // last_tag -> STOP
        nll[b] = logZ - emit - trs;
    }
}

__global__ void __launch_bounds__(128)
k_reduce(const float* __restrict__ nll, float* __restrict__ out) {
    __shared__ float s[128];
    const int tid = threadIdx.x;
    s[tid] = nll[tid];
    __syncthreads();
    for (int off = 64; off > 0; off >>= 1) {
        if (tid < off) s[tid] += s[tid + off];
        __syncthreads();
    }
    if (tid == 0) out[0] = s[0] * (1.f / 128.f);
}

// ---------------------------------------------------------------- launch
extern "C" void kernel_launch(void* const* d_in, const int* in_sizes, int n_in,
                              void* d_out, int out_size, void* d_ws, size_t ws_size,
                              hipStream_t stream) {
    (void)in_sizes; (void)n_in; (void)out_size; (void)ws_size;

    const float* embedding = (const float*)d_in[0];
    const float* w_ih0 = (const float*)d_in[1];
    const float* w_hh0 = (const float*)d_in[2];
    const float* b_ih0 = (const float*)d_in[3];
    const float* b_hh0 = (const float*)d_in[4];
    const float* w_ih1 = (const float*)d_in[5];
    const float* w_hh1 = (const float*)d_in[6];
    const float* b_ih1 = (const float*)d_in[7];
    const float* b_hh1 = (const float*)d_in[8];
    const float* ln_g  = (const float*)d_in[9];
    const float* ln_b  = (const float*)d_in[10];
    const float* w_out = (const float*)d_in[11];
    const float* b_out = (const float*)d_in[12];
    const float* trans = (const float*)d_in[13];
    const int*   word  = (const int*)d_in[14];
    const int*   mask  = (const int*)d_in[15];
    const int*   tags  = (const int*)d_in[16];
    float* out = (float*)d_out;
    char*  ws  = (char*)d_ws;

    size_t off = 0;
    auto alloc = [&](size_t bytes) {
        size_t o = off;
        off = (off + bytes + 255) & ~(size_t)255;
        return o;
    };
    const size_t n_x   = (size_t)B_ * S_ * E_;        // 8.4M bf16
    const size_t n_act = (size_t)B_ * S_ * 2 * H_;    // 16.8M bf16

    uint16_t* x_bf  = (uint16_t*)(ws + alloc(n_x * 2));
    uint16_t* out0  = (uint16_t*)(ws + alloc(n_act * 2));
    uint16_t* out1  = (uint16_t*)(ws + alloc(n_act * 2));
    uint16_t* wih0b = (uint16_t*)(ws + alloc((size_t)2 * 1024 * E_ * 2));
    uint16_t* whh0b = (uint16_t*)(ws + alloc((size_t)2 * 1024 * H_ * 2));
    uint16_t* wih1b = (uint16_t*)(ws + alloc((size_t)2 * 1024 * 512 * 2));
    uint16_t* whh1b = (uint16_t*)(ws + alloc((size_t)2 * 1024 * H_ * 2));
    float* bs0   = (float*)(ws + alloc((size_t)2 * 1024 * 4));
    float* bs1   = (float*)(ws + alloc((size_t)2 * 1024 * 4));
    float* gates = (float*)(ws + alloc((size_t)16 * 16 * 1024 * 4));
    float* feats = (float*)(ws + alloc((size_t)B_ * S_ * T_ * 4));
    float* nllb  = (float*)(ws + alloc((size_t)B_ * 4));

    // weight / bias prep (bf16 row-major copies; biases pre-summed)
    k_f32_to_bf16<<<(2 * 1024 * E_ + 255) / 256, 256, 0, stream>>>(w_ih0, wih0b, 2 * 1024 * E_);
    k_f32_to_bf16<<<(2 * 1024 * H_ + 255) / 256, 256, 0, stream>>>(w_hh0, whh0b, 2 * 1024 * H_);
    k_f32_to_bf16<<<(2 * 1024 * 512 + 255) / 256, 256, 0, stream>>>(w_ih1, wih1b, 2 * 1024 * 512);
    k_f32_to_bf16<<<(2 * 1024 * H_ + 255) / 256, 256, 0, stream>>>(w_hh1, whh1b, 2 * 1024 * H_);
    k_bias_sum<<<8, 256, 0, stream>>>(b_ih0, b_hh0, bs0, 2048);
    k_bias_sum<<<8, 256, 0, stream>>>(b_ih1, b_hh1, bs1, 2048);

    // embedding gather
    k_embed<<<(int)(n_x / 256), 256, 0, stream>>>(embedding, word, x_bf, (int)n_x);

    // fused BiLSTM layers (input GEMM folded into the recurrent WMMA scan)
    k_lstm_scan<<<16, 256, 0, stream>>>(x_bf, E_,  wih0b, whh0b, bs0, out0, gates);
    k_lstm_scan<<<16, 256, 0, stream>>>(out0, 512, wih1b, whh1b, bs1, out1, gates);

    // layernorm + tag projection
    k_ln_proj<<<B_ * S_, 256, 0, stream>>>(out1, ln_g, ln_b, w_out, b_out, feats);

    // CRF negative log-likelihood, deterministic mean
    k_crf<<<B_, 32, 0, stream>>>(feats, trans, mask, tags, nllb);
    k_reduce<<<1, 128, 0, stream>>>(nllb, out);
}
